// Glimpse_88776974008864
// MI455X (gfx1250) — matrix operand order repaired and checked by reading, not compiled, not run on hardware
//
#include <hip/hip_runtime.h>
#include <hip/hip_bf16.h>
#include <math.h>

// ---------------------------------------------------------------------------
// Glimpse kernel for MI455X (gfx1250, wave32, WMMA).
// Algebraically fused version:
//   Wq_eff[(m,g),d] = (1/8) * sum_h W_k_bar[m,h,d] * q[g,h]          (tiny)
//   scoresT[r,n]    = sum_d Wq_eff[r,d] * b[n,d]   (+mask -> -inf)   (WMMA GEMM 1)
//   softmax over n per row r (contiguous)                            (reduction)
//   attnT[r,n]      = exp(s - max)/sum  (f16)                        (elementwise)
//   T[r,d]          = sum_n attnT[r,n] * b[n,d]                      (WMMA GEMM 2)
//   q_s[m*64+h]     = sum_{g,d} W_v_bar[g,h,d] * T[m*16+g,d]         (small)
//   w2              = W_k^T (W_q q_s)                                (matvecs)
//   out[n]          = mask ? 10*tanh(dot(b[n],w2)/32) : -inf         (dot)
// b converted to f16 once (row-major b16 + transposed bT16); both WMMA GEMMs
// are software-pipelined (double-buffered fragments) so loads for K-step k+1
// overlap the WMMAs of K-step k.
// ---------------------------------------------------------------------------

typedef _Float16 h16_t __attribute__((ext_vector_type(16)));
typedef _Float16 h8_t  __attribute__((ext_vector_type(8)));
typedef float    f8_t  __attribute__((ext_vector_type(8)));

static constexpr int kD = 1024;   // feature dim
static constexpr int kN = 16384;  // sequence length
static constexpr int kR = 256;    // M*M = 16*16 score rows
static constexpr int kH = 64;     // head dim

#define WMMA_F16(A, B, C) \
  __builtin_amdgcn_wmma_f32_16x16x32_f16(false, (A), false, (B), (short)0, (C), false, false)

// A-matrix fragment (16x32 f16): lane holds row (lane&15); K chunks
// {0..7,16..23} for lanes 0-15 and {8..15,24..31} for lanes 16-31.
__device__ __forceinline__ h16_t load_a_frag(const _Float16* __restrict__ A, int lda,
                                             int row, int kb, int lane) {
  const int koff = (lane & 16) ? 8 : 0;
  const _Float16* p = A + (size_t)row * lda + kb + koff;
  h16_t f;
  ((h8_t*)&f)[0] = *(const h8_t*)(p);
  ((h8_t*)&f)[1] = *(const h8_t*)(p + 16);
  return f;
}

// B-matrix fragment (32x16 f16) from "B^T" storage BT[col][k] (k contiguous):
// lane holds column (lane&15), K chunks {0..7,16..23} for lanes 0-15 and {8..15,24..31} for lanes 16-31.
__device__ __forceinline__ h16_t load_bT_frag(const _Float16* __restrict__ BT, int ldb,
                                              int col0, int kb, int lane) {
  const int c = lane & 15;
  const int koff = (lane & 16) ? 8 : 0;
  const _Float16* p = BT + (size_t)(col0 + c) * ldb + kb + koff;
  h16_t f;
  ((h8_t*)&f)[0] = *(const h8_t*)(p);
  ((h8_t*)&f)[1] = *(const h8_t*)(p + 16);
  return f;
}

// ---------------------------------------------------------------------------
// Stage 1: Wq_eff_f16[r=(m*16+g)][d] = (1/8) * sum_h W_k_bar[m,h,d]*q[g,h]
__global__ void glimpse_wqeff(const float* __restrict__ Wkb,
                              const float* __restrict__ qsb,
                              _Float16* __restrict__ Wq16) {
  const int r = blockIdx.x;  // m*16+g
  const int m = r >> 4, g = r & 15;
  __shared__ float qsh[kH];
  if (threadIdx.x < kH) qsh[threadIdx.x] = qsb[g * kH + threadIdx.x];
  __syncthreads();
#pragma unroll
  for (int dd = 0; dd < 4; ++dd) {
    const int d = threadIdx.x + dd * 256;
    float s = 0.f;
#pragma unroll 4
    for (int h = 0; h < kH; ++h)
      s = fmaf(Wkb[(((size_t)(m * kH + h)) << 10) + d], qsh[h], s);
    Wq16[((size_t)r << 10) + d] = (_Float16)(s * 0.125f);
  }
}

// ---------------------------------------------------------------------------
// Stage 2: one pass over b producing BOTH f16 copies:
//   b16[n][d]  = (f16) b[n][d]   (row-major, for scores GEMM B-side)
//   bT16[d][n] = (f16) b[n][d]   (transposed, for T GEMM B-side)
__global__ void glimpse_convert_b(const float* __restrict__ b,
                                  _Float16* __restrict__ b16,
                                  _Float16* __restrict__ bT) {
  __shared__ _Float16 tile[16][17];
  const int t = threadIdx.x;
  const int n0 = blockIdx.x * 16, d0 = blockIdx.y * 16;
  const int i = t >> 4, j = t & 15;
  const _Float16 v = (_Float16)b[(size_t)(n0 + i) * kD + d0 + j];
  b16[(size_t)(n0 + i) * kD + d0 + j] = v;
  tile[i][j] = v;
  __syncthreads();
  bT[(size_t)(d0 + i) * kN + n0 + j] = tile[j][i];
}

// ---------------------------------------------------------------------------
// Stage 3: scoresT[r][n] = sum_d Wq16[r,d]*b16[n,d], mask -> -inf.
// One wave: 16 rows x 64 cols (4 WMMA tiles), K-loop over D, double-buffered.
__global__ void glimpse_gemm_scores(const _Float16* __restrict__ Wq16,
                                    const _Float16* __restrict__ b16,
                                    const int* __restrict__ mask,
                                    float* __restrict__ scoresT) {
  const int lane = threadIdx.x & 31;
  const int w = blockIdx.x * (blockDim.x >> 5) + (threadIdx.x >> 5);
  const int r0 = (w >> 8) * 16;   // 16 row tiles
  const int n0 = (w & 255) * 64;  // 256 col groups
  f8_t acc[4] = {};
  const int arow = r0 + (lane & 15);

  // prologue: fragments for kb = 0
  h16_t a  = load_a_frag(Wq16, kD, arow, 0, lane);
  h16_t b0 = load_bT_frag(b16, kD, n0 +  0, 0, lane);
  h16_t b1 = load_bT_frag(b16, kD, n0 + 16, 0, lane);
  h16_t b2 = load_bT_frag(b16, kD, n0 + 32, 0, lane);
  h16_t b3 = load_bT_frag(b16, kD, n0 + 48, 0, lane);

  for (int kb = 32; kb < kD; kb += 32) {
    // issue next K-step loads before consuming current fragments
    h16_t an  = load_a_frag(Wq16, kD, arow, kb, lane);
    h16_t c0 = load_bT_frag(b16, kD, n0 +  0, kb, lane);
    h16_t c1 = load_bT_frag(b16, kD, n0 + 16, kb, lane);
    h16_t c2 = load_bT_frag(b16, kD, n0 + 32, kb, lane);
    h16_t c3 = load_bT_frag(b16, kD, n0 + 48, kb, lane);
    acc[0] = WMMA_F16(a, b0, acc[0]);
    acc[1] = WMMA_F16(a, b1, acc[1]);
    acc[2] = WMMA_F16(a, b2, acc[2]);
    acc[3] = WMMA_F16(a, b3, acc[3]);
    a = an; b0 = c0; b1 = c1; b2 = c2; b3 = c3;
  }
  acc[0] = WMMA_F16(a, b0, acc[0]);
  acc[1] = WMMA_F16(a, b1, acc[1]);
  acc[2] = WMMA_F16(a, b2, acc[2]);
  acc[3] = WMMA_F16(a, b3, acc[3]);

  const int cn = lane & 15;
  const int rsel = (lane & 16) ? 8 : 0;
#pragma unroll
  for (int t = 0; t < 4; ++t) {
    const int n = n0 + t * 16 + cn;
    const bool mk = (mask[n] != 0);
#pragma unroll
    for (int j = 0; j < 8; ++j) {
      const float v = acc[t][j];
      scoresT[(size_t)(r0 + j + rsel) * kN + n] = mk ? v : -__builtin_inff();
    }
  }
}

// ---------------------------------------------------------------------------
// Stage 4: per-row (r) max and sum(exp) over n (contiguous rows of scoresT).
__global__ void glimpse_softmax_stats(const float* __restrict__ scoresT,
                                      float* __restrict__ rowmax,
                                      float* __restrict__ rowsum) {
  const int r = blockIdx.x;
  const float* row = scoresT + (size_t)r * kN;
  __shared__ float sm[256];
  float m = -__builtin_inff();
  for (int i = threadIdx.x; i < kN; i += 256) m = fmaxf(m, row[i]);
  sm[threadIdx.x] = m;
  __syncthreads();
  for (int st = 128; st > 0; st >>= 1) {
    if (threadIdx.x < st) sm[threadIdx.x] = fmaxf(sm[threadIdx.x], sm[threadIdx.x + st]);
    __syncthreads();
  }
  const float mx = sm[0];
  __syncthreads();
  float s = 0.f;
  for (int i = threadIdx.x; i < kN; i += 256) s += __expf(row[i] - mx);
  sm[threadIdx.x] = s;
  __syncthreads();
  for (int st = 128; st > 0; st >>= 1) {
    if (threadIdx.x < st) sm[threadIdx.x] += sm[threadIdx.x + st];
    __syncthreads();
  }
  if (threadIdx.x == 0) { rowmax[r] = mx; rowsum[r] = sm[0]; }
}

// ---------------------------------------------------------------------------
// Stage 5: attnT[r][n] = (f16) exp(scoresT - max[r]) / sum[r]
__global__ void glimpse_attn(const float* __restrict__ scoresT,
                             const float* __restrict__ rowmax,
                             const float* __restrict__ rowsum,
                             _Float16* __restrict__ attnT) {
  const size_t idx = (size_t)blockIdx.x * 256 + threadIdx.x;
  const int r = (int)(idx >> 14);  // / kN
  const float v = __expf(scoresT[idx] - rowmax[r]) * (1.f / rowsum[r]);
  attnT[idx] = (_Float16)v;
}

// ---------------------------------------------------------------------------
// Stage 6: T[r][d] = sum_n attnT[r,n] * b[n,d]  (uses bT16). K = 16384.
__global__ void glimpse_gemm_T(const _Float16* __restrict__ attnT,
                               const _Float16* __restrict__ bT16,
                               float* __restrict__ T) {
  const int lane = threadIdx.x & 31;
  const int w = blockIdx.x * (blockDim.x >> 5) + (threadIdx.x >> 5);
  const int r0 = (w >> 4) * 16;  // 16 row tiles
  const int d0 = (w & 15) * 64;  // 16 col groups
  f8_t acc[4] = {};
  const int arow = r0 + (lane & 15);

  h16_t a  = load_a_frag(attnT, kN, arow, 0, lane);
  h16_t b0 = load_bT_frag(bT16, kN, d0 +  0, 0, lane);
  h16_t b1 = load_bT_frag(bT16, kN, d0 + 16, 0, lane);
  h16_t b2 = load_bT_frag(bT16, kN, d0 + 32, 0, lane);
  h16_t b3 = load_bT_frag(bT16, kN, d0 + 48, 0, lane);

  for (int kb = 32; kb < kN; kb += 32) {
    h16_t an  = load_a_frag(attnT, kN, arow, kb, lane);
    h16_t c0 = load_bT_frag(bT16, kN, d0 +  0, kb, lane);
    h16_t c1 = load_bT_frag(bT16, kN, d0 + 16, kb, lane);
    h16_t c2 = load_bT_frag(bT16, kN, d0 + 32, kb, lane);
    h16_t c3 = load_bT_frag(bT16, kN, d0 + 48, kb, lane);
    acc[0] = WMMA_F16(a, b0, acc[0]);
    acc[1] = WMMA_F16(a, b1, acc[1]);
    acc[2] = WMMA_F16(a, b2, acc[2]);
    acc[3] = WMMA_F16(a, b3, acc[3]);
    a = an; b0 = c0; b1 = c1; b2 = c2; b3 = c3;
  }
  acc[0] = WMMA_F16(a, b0, acc[0]);
  acc[1] = WMMA_F16(a, b1, acc[1]);
  acc[2] = WMMA_F16(a, b2, acc[2]);
  acc[3] = WMMA_F16(a, b3, acc[3]);

  const int cn = lane & 15;
  const int rsel = (lane & 16) ? 8 : 0;
#pragma unroll
  for (int t = 0; t < 4; ++t) {
#pragma unroll
    for (int j = 0; j < 8; ++j)
      T[(size_t)(r0 + j + rsel) * kD + d0 + t * 16 + cn] = acc[t][j];
  }
}

// ---------------------------------------------------------------------------
// Stage 7: q_s[m*64+h] = sum_{g,d} W_v_bar[g,h,d] * T[m*16+g, d]
__global__ void glimpse_qs(const float* __restrict__ Wv,
                           const float* __restrict__ T,
                           float* __restrict__ qs) {
  const int mh = blockIdx.x;  // m*64 + h
  const int m = mh >> 6, h = mh & 63;
  float s = 0.f;
  for (int idx = threadIdx.x; idx < 16 * kD; idx += 256) {
    const int g = idx >> 10, d = idx & 1023;
    s = fmaf(Wv[(((size_t)(g * kH + h)) << 10) + d],
             T[(((size_t)(m * 16 + g)) << 10) + d], s);
  }
  __shared__ float sm[256];
  sm[threadIdx.x] = s;
  __syncthreads();
  for (int st = 128; st > 0; st >>= 1) {
    if (threadIdx.x < st) sm[threadIdx.x] += sm[threadIdx.x + st];
    __syncthreads();
  }
  if (threadIdx.x == 0) qs[mh] = sm[0];
}

// y[i] = sum_j A[i,j] * x[j]   (A row-major 1024x1024)
__global__ void glimpse_matvec_row(const float* __restrict__ A,
                                   const float* __restrict__ x,
                                   float* __restrict__ y) {
  const int i = blockIdx.x;
  float s = 0.f;
  for (int j = threadIdx.x; j < kD; j += 256)
    s = fmaf(A[((size_t)i << 10) + j], x[j], s);
  __shared__ float sm[256];
  sm[threadIdx.x] = s;
  __syncthreads();
  for (int st = 128; st > 0; st >>= 1) {
    if (threadIdx.x < st) sm[threadIdx.x] += sm[threadIdx.x + st];
    __syncthreads();
  }
  if (threadIdx.x == 0) y[i] = sm[0];
}

// y[e] = sum_d A[d,e] * x[d]   (A^T matvec)
__global__ void glimpse_matvec_col(const float* __restrict__ A,
                                   const float* __restrict__ x,
                                   float* __restrict__ y) {
  const int e = blockIdx.x;
  float s = 0.f;
  for (int d = threadIdx.x; d < kD; d += 256)
    s = fmaf(A[((size_t)d << 10) + e], x[d], s);
  __shared__ float sm[256];
  sm[threadIdx.x] = s;
  __syncthreads();
  for (int st = 128; st > 0; st >>= 1) {
    if (threadIdx.x < st) sm[threadIdx.x] += sm[threadIdx.x + st];
    __syncthreads();
  }
  if (threadIdx.x == 0) y[e] = sm[0];
}

// ---------------------------------------------------------------------------
// Stage 8: out[n] = mask ? 10*tanh(dot(b[n],w2)/32) : -inf
__global__ void glimpse_final(const float* __restrict__ b,
                              const float* __restrict__ w2,
                              const int* __restrict__ mask,
                              float* __restrict__ out) {
  const int n = blockIdx.x;
  float s = 0.f;
  for (int d = threadIdx.x; d < kD; d += 256)
    s = fmaf(b[((size_t)n << 10) + d], w2[d], s);
  __shared__ float sm[256];
  sm[threadIdx.x] = s;
  __syncthreads();
  for (int st = 128; st > 0; st >>= 1) {
    if (threadIdx.x < st) sm[threadIdx.x] += sm[threadIdx.x + st];
    __syncthreads();
  }
  if (threadIdx.x == 0)
    out[n] = mask[n] ? 10.f * tanhf(sm[0] * (1.f / 32.f)) : -__builtin_inff();
}

// ---------------------------------------------------------------------------
extern "C" void kernel_launch(void* const* d_in, const int* in_sizes, int n_in,
                              void* d_out, int out_size, void* d_ws, size_t ws_size,
                              hipStream_t stream) {
  const float* b    = (const float*)d_in[0];  // (N, D)
  const float* qsb  = (const float*)d_in[1];  // (D,)
  const int*   mask = (const int*)d_in[2];    // (N,)
  const float* Wkb  = (const float*)d_in[3];  // (M, H, D)
  const float* Wv   = (const float*)d_in[4];  // (M, H, D)
  const float* Wk   = (const float*)d_in[5];  // (D, D)
  const float* Wq   = (const float*)d_in[6];  // (D, D)
  float* out = (float*)d_out;

  char* ws = (char*)d_ws;
  size_t off = 0;
  auto alloc = [&](size_t bytes) { char* p = ws + off; off += (bytes + 255) & ~(size_t)255; return p; };
  _Float16* Wq16    = (_Float16*)alloc((size_t)kR * kD * 2);   // 512 KB
  _Float16* b16     = (_Float16*)alloc((size_t)kN * kD * 2);   // 33.5 MB
  _Float16* bT16    = (_Float16*)alloc((size_t)kD * kN * 2);   // 33.5 MB
  float*    scoresT = (float*)   alloc((size_t)kR * kN * 4);   // 16.8 MB
  _Float16* attnT   = (_Float16*)alloc((size_t)kR * kN * 2);   // 8.4 MB
  float*    T       = (float*)   alloc((size_t)kR * kD * 4);   // 1 MB
  float*    rowmax  = (float*)   alloc(kR * 4);
  float*    rowsum  = (float*)   alloc(kR * 4);
  float*    qs      = (float*)   alloc(kD * 4);
  float*    w1      = (float*)   alloc(kD * 4);
  float*    w2      = (float*)   alloc(kD * 4);
  (void)ws_size; (void)in_sizes; (void)n_in; (void)out_size;

  glimpse_wqeff<<<kR, 256, 0, stream>>>(Wkb, qsb, Wq16);
  glimpse_convert_b<<<dim3(kN / 16, kD / 16), 256, 0, stream>>>(b, b16, bT16);
  // 4096 waves = 16 row-tiles x 256 col-groups; 8 waves per block.
  glimpse_gemm_scores<<<512, 256, 0, stream>>>(Wq16, b16, mask, scoresT);
  glimpse_softmax_stats<<<kR, 256, 0, stream>>>(scoresT, rowmax, rowsum);
  glimpse_attn<<<(kR * kN) / 256, 256, 0, stream>>>(scoresT, rowmax, rowsum, attnT);
  // 256 waves = 16 row-tiles x 16 col-groups; 8 waves per block.
  glimpse_gemm_T<<<32, 256, 0, stream>>>(attnT, bT16, T);
  glimpse_qs<<<kD, 256, 0, stream>>>(Wv, T, qs);
  glimpse_matvec_row<<<kD, 256, 0, stream>>>(Wq, qs, w1);   // w1 = W_q @ q_s
  glimpse_matvec_col<<<kD, 256, 0, stream>>>(Wk, w1, w2);   // w2 = W_k^T @ w1
  glimpse_final<<<kN, 256, 0, stream>>>(b, w2, mask, out);
}
